// MOORENetwork_39006892982624
// MI455X (gfx1250) — matrix operand order repaired and checked by reading, not compiled, not run on hardware
//
#include <hip/hip_runtime.h>

// ---------------------------------------------------------------------------
// MOORE network, MI455X (gfx1250) implementation.
// bf16 WMMA (v_wmma_f32_16x16x32_bf16) for the 3 batched expert GEMMs,
// fp32 Gram-Schmidt + head fused in one block-per-row kernel.
// ---------------------------------------------------------------------------

typedef __bf16 bf16;
typedef bf16  v8bf  __attribute__((ext_vector_type(8)));
typedef bf16  v16bf __attribute__((ext_vector_type(16)));
typedef float v8f   __attribute__((ext_vector_type(8)));

static constexpr int cB   = 4096;
static constexpr int cOBS = 512;
static constexpr int cT   = 10;
static constexpr int cE   = 8;
static constexpr int cW   = 1024;
static constexpr int cH   = 12;
static constexpr int cX   = cOBS + cT;   // 522

// ---------------------------------------------------------------------------
// obs slice of x -> bf16 (B x OBS)
// ---------------------------------------------------------------------------
__global__ __launch_bounds__(256)
void obs_to_bf16(const float* __restrict__ x, bf16* __restrict__ obs) {
  int i = blockIdx.x * blockDim.x + threadIdx.x;
  if (i < cB * cOBS) {
    int b = i / cOBS, k = i - b * cOBS;
    obs[i] = (bf16)x[(size_t)b * cX + k];
  }
}

// ---------------------------------------------------------------------------
// Per-expert weight (K x N) fp32 -> transposed (N x K) bf16.
// Transposed layout makes the WMMA B-fragment two contiguous 16B runs per lane.
// ---------------------------------------------------------------------------
__global__ __launch_bounds__(256)
void transpose_to_bf16(const float* __restrict__ in, bf16* __restrict__ out,
                       int K, int N) {
  int e = blockIdx.z;
  size_t plane = (size_t)K * N;
  for (size_t i = (size_t)blockIdx.x * blockDim.x + threadIdx.x; i < plane;
       i += (size_t)gridDim.x * blockDim.x) {
    size_t n = i / K, k = i - n * K;
    out[e * plane + i] = (bf16)in[e * plane + k * N + n];
  }
}

// ---------------------------------------------------------------------------
// Batched-expert GEMM via v_wmma_f32_16x16x32_bf16 (wave32).
// Each wave: 16(M) x 64(N) tile (4 accumulators, A reused 4x per k-step).
// Block: 4 waves -> 64x64 tile. Grid: (N/64, M/64, E).
//   A : activations, row-major (M x K) bf16, per-expert stride a_estride
//   Bt: weights, transposed (N x K) bf16, per-expert plane N*K
//   fragment layouts follow CDNA5 ISA 7.12.2:
//     A lane<16: M=lane,    elems = K {k0..k0+7, k0+16..k0+23}
//       lane>=16:M=lane-16, elems = K {k0+8..15, k0+24..31}
//     B lane<16: N=lane,    elems = K {k0..k0+7, k0+16..k0+23}
//       lane>=16:N=lane-16, elems = K {k0+8..15, k0+24..31}
//     C vgpr r:  lanes0-15 -> M=r, lanes16-31 -> M=r+8; N = lane&15
// ---------------------------------------------------------------------------
template <bool RELU, bool BF16OUT>
__global__ __launch_bounds__(128)
void gemm_wmma_bf16(const bf16* __restrict__ A, size_t a_estride,
                    const bf16* __restrict__ Bt,
                    const float* __restrict__ bias,
                    void* __restrict__ Cout,
                    int M, int K, int N) {
  const int lane = threadIdx.x;          // 0..31
  const int wave = threadIdx.y;          // 0..3
  const int e    = blockIdx.z;
  const int m0   = blockIdx.y * 64 + wave * 16;
  const int n0   = blockIdx.x * 64;
  const int lm   = lane & 15;
  const bool hi  = lane >= 16;

  const bf16* Ae = A  + (size_t)e * a_estride;
  const bf16* Be = Bt + (size_t)e * (size_t)N * K;

  const bf16* arow = Ae + (size_t)(m0 + lm) * K + (hi ? 8 : 0);
  const bf16* brow[4];
#pragma unroll
  for (int i = 0; i < 4; ++i)
    brow[i] = Be + (size_t)(n0 + 16 * i + lm) * K + (hi ? 8 : 0);

  float bv[4];
#pragma unroll
  for (int i = 0; i < 4; ++i)
    bv[i] = bias[(size_t)e * N + n0 + 16 * i + lm];

  v8f c0 = {}, c1 = {}, c2 = {}, c3 = {};

  for (int k0 = 0; k0 < K; k0 += 32) {
    v8bf alo = *(const v8bf*)(arow + k0);
    v8bf ahi = *(const v8bf*)(arow + k0 + 16);
    v16bf a = __builtin_shufflevector(alo, ahi, 0, 1, 2, 3, 4, 5, 6, 7,
                                      8, 9, 10, 11, 12, 13, 14, 15);
    v16bf b[4];
#pragma unroll
    for (int i = 0; i < 4; ++i) {
      v8bf blo = *(const v8bf*)(brow[i] + k0);
      v8bf bhi = *(const v8bf*)(brow[i] + k0 + 16);
      b[i] = __builtin_shufflevector(blo, bhi, 0, 1, 2, 3, 4, 5, 6, 7,
                                     8, 9, 10, 11, 12, 13, 14, 15);
    }
    c0 = __builtin_amdgcn_wmma_f32_16x16x32_bf16(false, a, false, b[0],
                                                 (short)0, c0, false, false);
    c1 = __builtin_amdgcn_wmma_f32_16x16x32_bf16(false, a, false, b[1],
                                                 (short)0, c1, false, false);
    c2 = __builtin_amdgcn_wmma_f32_16x16x32_bf16(false, a, false, b[2],
                                                 (short)0, c2, false, false);
    c3 = __builtin_amdgcn_wmma_f32_16x16x32_bf16(false, a, false, b[3],
                                                 (short)0, c3, false, false);
  }

  v8f cc[4] = {c0, c1, c2, c3};
  const size_t cbase = (size_t)e * (size_t)M * N;
#pragma unroll
  for (int i = 0; i < 4; ++i) {
    const int n = n0 + 16 * i + lm;
#pragma unroll
    for (int r = 0; r < 8; ++r) {
      const int m = m0 + r + (hi ? 8 : 0);
      float v = cc[i][r] + bv[i];
      if (RELU) v = v > 0.f ? v : 0.f;
      if (BF16OUT)
        ((bf16*)Cout)[cbase + (size_t)m * N + n] = (bf16)v;
      else
        ((float*)Cout)[cbase + (size_t)m * N + n] = v;
    }
  }
}

// ---------------------------------------------------------------------------
// Fused Gram-Schmidt + tanh feature mix + per-task head. One block per row.
// ---------------------------------------------------------------------------
__device__ __forceinline__ float block_reduce(float val, float* red, int t) {
  red[t] = val;
  __syncthreads();
#pragma unroll
  for (int s = 128; s > 0; s >>= 1) {
    if (t < s) red[t] += red[t + s];
    __syncthreads();
  }
  float r = red[0];
  __syncthreads();
  return r;
}

__global__ __launch_bounds__(256)
void gs_head(const float* __restrict__ eo, const float* __restrict__ x,
             const float* __restrict__ te_W, const float* __restrict__ head_W,
             const float* __restrict__ head_b, float* __restrict__ out) {
  __shared__ float v[cE][cW];     // 32 KB, orthonormalized in place
  __shared__ float feat[cW];      // 4 KB
  __shared__ float red[256];
  __shared__ float accum[cH];
  __shared__ int   s_idx;

  const int t = threadIdx.x;
  const int b = blockIdx.x;

  // load eo[b] for all experts; eo layout (E, B, W)
  for (int i = t; i < cE * cW; i += 256) {
    int e = i >> 10, k = i & (cW - 1);
    v[e][k] = eo[((size_t)e * cB + b) * cW + k];
  }
  if (t == 0) {
    const float* oh = x + (size_t)b * cX + cOBS;
    int best = 0; float bmax = oh[0];
#pragma unroll
    for (int j = 1; j < cT; ++j)
      if (oh[j] > bmax) { bmax = oh[j]; best = j; }
    s_idx = best;
  }
  if (t < cH) accum[t] = 0.f;
  __syncthreads();

  // classical Gram-Schmidt (coeffs vs. original vector, like the reference)
  for (int i = 0; i < cE; ++i) {
    float cj[cE];
    for (int j = 0; j < i; ++j) {
      float p = 0.f;
      for (int k = t; k < cW; k += 256) p += v[i][k] * v[j][k];
      cj[j] = block_reduce(p, red, t);
    }
    if (i > 0) {
      for (int k = t; k < cW; k += 256) {
        float acc = v[i][k];
        for (int j = 0; j < i; ++j) acc -= cj[j] * v[j][k];
        v[i][k] = acc;
      }
      __syncthreads();
    }
    float ss = 0.f;
    for (int k = t; k < cW; k += 256) ss += v[i][k] * v[i][k];
    float inv = 1.0f / (sqrtf(block_reduce(ss, red, t)) + 1e-8f);
    for (int k = t; k < cW; k += 256) v[i][k] *= inv;
    __syncthreads();
  }

  // feat = tanh(sum_e te[e] * basis[e])
  const int idx = s_idx;
  float te[cE];
#pragma unroll
  for (int e = 0; e < cE; ++e) te[e] = te_W[idx * cE + e];
  for (int k = t; k < cW; k += 256) {
    float s = 0.f;
#pragma unroll
    for (int e = 0; e < cE; ++e) s += v[e][k] * te[e];
    feat[k] = tanhf(s);
  }
  __syncthreads();

  // head: feat(1024) @ head_W[idx](1024x12) + head_b[idx]
  float part[cH];
#pragma unroll
  for (int h = 0; h < cH; ++h) part[h] = 0.f;
  for (int k = t; k < cW; k += 256) {
    float f = feat[k];
    const float* hw = head_W + ((size_t)idx * cW + k) * cH;
#pragma unroll
    for (int h = 0; h < cH; ++h) part[h] += f * hw[h];
  }
#pragma unroll
  for (int h = 0; h < cH; ++h) atomicAdd(&accum[h], part[h]);
  __syncthreads();
  if (t < cH) out[(size_t)b * cH + t] = accum[t] + head_b[idx * cH + t];
}

// ---------------------------------------------------------------------------
// launch
// ---------------------------------------------------------------------------
extern "C" void kernel_launch(void* const* d_in, const int* in_sizes, int n_in,
                              void* d_out, int out_size, void* d_ws,
                              size_t ws_size, hipStream_t stream) {
  (void)in_sizes; (void)n_in; (void)out_size; (void)ws_size;
  const float* x      = (const float*)d_in[0];
  const float* te_W   = (const float*)d_in[1];
  const float* W0     = (const float*)d_in[2];
  const float* b0     = (const float*)d_in[3];
  const float* W1     = (const float*)d_in[4];
  const float* b1     = (const float*)d_in[5];
  const float* W2     = (const float*)d_in[6];
  const float* b2     = (const float*)d_in[7];
  const float* head_W = (const float*)d_in[8];
  const float* head_b = (const float*)d_in[9];
  float* out = (float*)d_out;

  char* ws = (char*)d_ws;
  size_t off = 0;
  auto alloc = [&](size_t bytes) -> void* {
    void* p = ws + off;
    off = (off + bytes + 255) & ~(size_t)255;
    return p;
  };
  bf16*  obs_bf = (bf16*)alloc((size_t)cB * cOBS * 2);
  bf16*  W0t    = (bf16*)alloc((size_t)cE * cW * cOBS * 2);
  bf16*  W1t    = (bf16*)alloc((size_t)cE * cW * cW * 2);
  bf16*  W2t    = (bf16*)alloc((size_t)cE * cW * cW * 2);
  bf16*  h1     = (bf16*)alloc((size_t)cE * cB * cW * 2);
  bf16*  h2     = (bf16*)alloc((size_t)cE * cB * cW * 2);
  float* eo     = (float*)alloc((size_t)cE * cB * cW * 4);

  // preprocess
  obs_to_bf16<<<(cB * cOBS + 255) / 256, 256, 0, stream>>>(x, obs_bf);
  transpose_to_bf16<<<dim3(512, 1, cE), 256, 0, stream>>>(W0, W0t, cOBS, cW);
  transpose_to_bf16<<<dim3(1024, 1, cE), 256, 0, stream>>>(W1, W1t, cW, cW);
  transpose_to_bf16<<<dim3(1024, 1, cE), 256, 0, stream>>>(W2, W2t, cW, cW);

  // expert MLP: three batched WMMA GEMMs
  dim3 blk(32, 4);
  gemm_wmma_bf16<true, true><<<dim3(cW / 64, cB / 64, cE), blk, 0, stream>>>(
      obs_bf, (size_t)0, W0t, b0, h1, cB, cOBS, cW);
  gemm_wmma_bf16<true, true><<<dim3(cW / 64, cB / 64, cE), blk, 0, stream>>>(
      h1, (size_t)cB * cW, W1t, b1, h2, cB, cW, cW);
  gemm_wmma_bf16<false, false><<<dim3(cW / 64, cB / 64, cE), blk, 0, stream>>>(
      h2, (size_t)cB * cW, W2t, b2, eo, cB, cW, cW);

  // Gram-Schmidt + tanh mix + per-task head
  gs_head<<<cB, 256, 0, stream>>>(eo, x, te_W, head_W, head_b, out);
}
